// DirGCNConv_21165598835486
// MI455X (gfx1250) — 8-bit matrix operand order repaired and checked by reading, compiled, not run on hardware
//
#include <hip/hip_runtime.h>
#include <hip/hip_bf16.h>

typedef __attribute__((ext_vector_type(8))) int   v8i;
typedef __attribute__((ext_vector_type(8))) float v8f;
typedef __attribute__((ext_vector_type(2))) float v2f;

#define DIMF 256

__global__ __launch_bounds__(256) void scatter_edges(const int* __restrict__ ei, int E, int N,
                                                     unsigned char* __restrict__ A,
                                                     unsigned char* __restrict__ At) {
    int e = blockIdx.x * blockDim.x + threadIdx.x;
    if (e >= E) return;
    int s = ei[e];
    int d = ei[E + e];
    A [(size_t)s * N + d] = 1;
    At[(size_t)d * N + s] = 1;
}

__device__ __forceinline__ int sum_bytes(unsigned int w) {
    return (int)(w & 255u) + (int)((w >> 8) & 255u) + (int)((w >> 16) & 255u) + (int)(w >> 24);
}

__global__ __launch_bounds__(256) void rownorm_u8(const unsigned char* __restrict__ X, int N,
                                                  float* __restrict__ nrm) {
    const int lane = threadIdx.x & 31;
    const int row  = blockIdx.x * 8 + (threadIdx.x >> 5);
    const uint4* p = (const uint4*)(X + (size_t)row * N);
    int acc = 0;
    for (int i = lane; i < N / 16; i += 32) {
        uint4 w = p[i];
        acc += sum_bytes(w.x) + sum_bytes(w.y) + sum_bytes(w.z) + sum_bytes(w.w);
    }
    for (int off = 16; off; off >>= 1) acc += __shfl_xor(acc, off, 32);
    if (lane == 0) nrm[row] = acc > 0 ? rsqrtf((float)acc) : 0.0f;
}

__global__ __launch_bounds__(256) void rownorm_i32(const int* __restrict__ C, int N,
                                                   float* __restrict__ nrm) {
    const int lane = threadIdx.x & 31;
    const int row  = blockIdx.x * 8 + (threadIdx.x >> 5);
    const int4* p = (const int4*)(C + (size_t)row * N);
    int acc = 0;
    for (int i = lane; i < N / 4; i += 32) {
        int4 w = p[i];
        acc += w.x + w.y + w.z + w.w;
    }
    for (int off = 16; off; off >>= 1) acc += __shfl_xor(acc, off, 32);
    if (lane == 0) nrm[row] = acc > 0 ? rsqrtf((float)acc) : 0.0f;
}

__global__ __launch_bounds__(256) void colnorm_i32(const int* __restrict__ C, int N,
                                                   float* __restrict__ nrm) {
    const int j = blockIdx.x * blockDim.x + threadIdx.x;
    int acc = 0;
    for (int r = 0; r < N; ++r) acc += C[(size_t)r * N + j];
    nrm[j] = acc > 0 ? rsqrtf((float)acc) : 0.0f;
}

__global__ void bias_init(const float* __restrict__ bs, const float* __restrict__ bd,
                          float* __restrict__ out) {
    out[(size_t)blockIdx.x * DIMF + threadIdx.x] =
        0.5f * (bs[threadIdx.x] + bd[threadIdx.x]);
}

__global__ __launch_bounds__(128) void gram_iu8(const unsigned char* __restrict__ X,
                                                int* __restrict__ C, int N) {
    const int lane = threadIdx.x & 31;
    const int wave = threadIdx.x >> 5;
    const int g    = lane >> 4;
    const int r16  = lane & 15;
    const int m0   = blockIdx.y * 64 + wave * 16;
    const int n0   = blockIdx.x * 64;

    const unsigned char* rowA = X + (size_t)(m0 + r16) * N;
    const unsigned char* rowB0 = X + (size_t)(n0 +  0 + r16) * N + 16 * g;
    const unsigned char* rowB1 = X + (size_t)(n0 + 16 + r16) * N + 16 * g;
    const unsigned char* rowB2 = X + (size_t)(n0 + 32 + r16) * N + 16 * g;
    const unsigned char* rowB3 = X + (size_t)(n0 + 48 + r16) * N + 16 * g;

    v8i acc[4] = {v8i{}, v8i{}, v8i{}, v8i{}};

    for (int k0 = 0; k0 < N; k0 += 64) {
        __builtin_prefetch(rowA + k0 + 512, 0, 1);
        uint2 a01 = *(const uint2*)(rowA + k0 +      16 * g);
        uint2 a23 = *(const uint2*)(rowA + k0 +  8 + 16 * g);
        uint2 a45 = *(const uint2*)(rowA + k0 + 32 + 16 * g);
        uint2 a67 = *(const uint2*)(rowA + k0 + 40 + 16 * g);
        uint4 b0[4], b1[4];
        b0[0] = *(const uint4*)(rowB0 + k0);  b1[0] = *(const uint4*)(rowB0 + k0 + 32);
        b0[1] = *(const uint4*)(rowB1 + k0);  b1[1] = *(const uint4*)(rowB1 + k0 + 32);
        b0[2] = *(const uint4*)(rowB2 + k0);  b1[2] = *(const uint4*)(rowB2 + k0 + 32);
        b0[3] = *(const uint4*)(rowB3 + k0);  b1[3] = *(const uint4*)(rowB3 + k0 + 32);

        v8i a = {(int)a01.x, (int)a01.y, (int)a23.x, (int)a23.y,
                 (int)a45.x, (int)a45.y, (int)a67.x, (int)a67.y};
#pragma unroll
        for (int s = 0; s < 4; ++s) {
            v8i b = {(int)b0[s].x, (int)b0[s].y, (int)b0[s].z, (int)b0[s].w,
                     (int)b1[s].x, (int)b1[s].y, (int)b1[s].z, (int)b1[s].w};
            acc[s] = __builtin_amdgcn_wmma_i32_16x16x64_iu8(
                false, a, false, b, acc[s], false, false);
        }
    }

#pragma unroll
    for (int s = 0; s < 4; ++s) {
        const int n = n0 + s * 16 + r16;
#pragma unroll
        for (int v = 0; v < 8; ++v) {
            const int m = m0 + v + 8 * g;
            int val = acc[s][v];
            if (m == n || X[(size_t)m * N + n] != 0) val = 0;
            C[(size_t)m * N + n] = val;
        }
    }
}

__global__ __launch_bounds__(128) void gemm_xw(const float* __restrict__ x,
                                               const float* __restrict__ W,
                                               float* __restrict__ y, int K) {
    const int lane = threadIdx.x & 31;
    const int wave = threadIdx.x >> 5;
    const int g    = lane >> 4;
    const int r16  = lane & 15;
    const int m0   = blockIdx.y * 64 + wave * 16;
    const int n0   = blockIdx.x * 16;

    const float* pa = x + (size_t)(m0 + r16) * K + 2 * g;
    const float* pb = W + (size_t)(n0 + r16) * K + 2 * g;
    v8f c = {};
    for (int k0 = 0; k0 < K; k0 += 4) {
        v2f a = *(const v2f*)(pa + k0);
        v2f b = *(const v2f*)(pb + k0);
        c = __builtin_amdgcn_wmma_f32_16x16x4_f32(false, a, false, b, (short)0, c, false, false);
    }
#pragma unroll
    for (int v = 0; v < 8; ++v)
        y[(size_t)(m0 + v + 8 * g) * DIMF + n0 + r16] = c[v];
}

template <typename T>
__global__ __launch_bounds__(256) void spmm_acc(const T* __restrict__ Mt,
                                                const float* __restrict__ rs,
                                                const float* __restrict__ cs,
                                                const float* __restrict__ xw,
                                                float* __restrict__ out,
                                                float coef, int N) {
    __shared__ float s[4][256];
    const int t  = threadIdx.x;
    const int i0 = blockIdx.x * 4;
    float acc[4] = {0.f, 0.f, 0.f, 0.f};
    for (int j0 = 0; j0 < N; j0 += 256) {
        const float cw = cs[j0 + t];
#pragma unroll
        for (int r = 0; r < 4; ++r)
            s[r][t] = (float)Mt[(size_t)(i0 + r) * N + j0 + t] * cw;
        __syncthreads();
        for (int jj = 0; jj < 256; ++jj) {
            const float xv = xw[(size_t)(j0 + jj) * DIMF + t];
#pragma unroll
            for (int r = 0; r < 4; ++r) acc[r] += s[r][jj] * xv;
        }
        __syncthreads();
    }
#pragma unroll
    for (int r = 0; r < 4; ++r)
        out[(size_t)(i0 + r) * DIMF + t] += coef * rs[i0 + r] * acc[r];
}

extern "C" void kernel_launch(void* const* d_in, const int* in_sizes, int n_in,
                              void* d_out, int out_size, void* d_ws, size_t ws_size,
                              hipStream_t stream) {
    const float* x    = (const float*)d_in[0];
    const int*   ei   = (const int*)  d_in[1];
    const float* Wsrc = (const float*)d_in[2];
    const float* bsrc = (const float*)d_in[3];
    const float* Wdst = (const float*)d_in[4];
    const float* bdst = (const float*)d_in[5];
    float* out = (float*)d_out;

    const int D = DIMF;
    const int N = in_sizes[0] / D;
    const int E = in_sizes[1] / 2;

    char* p = (char*)d_ws;
    unsigned char* A   = (unsigned char*)p; p += (size_t)N * N;
    unsigned char* At  = (unsigned char*)p; p += (size_t)N * N;
    int*   C   = (int*)p;   p += (size_t)N * N * sizeof(int);
    float* xws = (float*)p; p += (size_t)N * D * sizeof(float);
    float* xwd = (float*)p; p += (size_t)N * D * sizeof(float);
    float* o1  = (float*)p; p += (size_t)N * sizeof(float);
    float* i1  = (float*)p; p += (size_t)N * sizeof(float);
    float* r2  = (float*)p; p += (size_t)N * sizeof(float);
    float* c2  = (float*)p; p += (size_t)N * sizeof(float);

    hipMemsetAsync(A,  0, (size_t)N * N, stream);
    hipMemsetAsync(At, 0, (size_t)N * N, stream);
    scatter_edges<<<(E + 255) / 256, 256, 0, stream>>>(ei, E, N, A, At);
    rownorm_u8<<<N / 8, 256, 0, stream>>>(A,  N, o1);
    rownorm_u8<<<N / 8, 256, 0, stream>>>(At, N, i1);

    bias_init<<<N, 256, 0, stream>>>(bsrc, bdst, out);
    gemm_xw<<<dim3(D / 16, N / 64), 128, 0, stream>>>(x, Wsrc, xws, D);
    gemm_xw<<<dim3(D / 16, N / 64), 128, 0, stream>>>(x, Wdst, xwd, D);

    spmm_acc<unsigned char><<<N / 4, 256, 0, stream>>>(A,  o1, i1, xws, out, 0.35f, N);
    spmm_acc<unsigned char><<<N / 4, 256, 0, stream>>>(At, i1, o1, xwd, out, 0.35f, N);

    gram_iu8<<<dim3(N / 64, N / 64), 128, 0, stream>>>(A, C, N);
    rownorm_i32<<<N / 8,   256, 0, stream>>>(C, N, r2);
    colnorm_i32<<<N / 256, 256, 0, stream>>>(C, N, c2);
    spmm_acc<int><<<N / 4, 256, 0, stream>>>(C, r2, c2, xws, out, 0.15f, N);

    gram_iu8<<<dim3(N / 64, N / 64), 128, 0, stream>>>(At, C, N);
    rownorm_i32<<<N / 8,   256, 0, stream>>>(C, N, r2);
    colnorm_i32<<<N / 256, 256, 0, stream>>>(C, N, c2);
    spmm_acc<int><<<N / 4, 256, 0, stream>>>(C, r2, c2, xwd, out, 0.15f, N);
}
